// RBMSites_4776003633308
// MI455X (gfx1250) — matrix operand order repaired and checked by reading, not compiled, not run on hardware
//
#include <hip/hip_runtime.h>
#include <stdint.h>

#define SORB   64
#define HID    256
#define BATCH  8192
#define NSTEP  32
#define SPLIT  4
#define STEPS_PER (NSTEP/SPLIT)

typedef __attribute__((ext_vector_type(16))) _Float16 v16h;
typedef __attribute__((ext_vector_type(8)))  float    v8f;

// Per-step slab: 256x64 f16 padded weights (32KB) + 256 float2 site weights (2KB)
#define SLAB_BYTES 34816
#define SLAB_VEC4  2176   // SLAB_BYTES / 16

// ---- CDNA5 async global->LDS path (guarded: falls back to reg-staged copy) ----
#if defined(__gfx1250__) && __has_builtin(__builtin_amdgcn_global_load_async_to_lds_b128) && __has_builtin(__builtin_amdgcn_s_wait_asynccnt)
#define USE_ASYNC_LDS 1
typedef int i32x4 __attribute__((vector_size(16)));
typedef __attribute__((address_space(1))) i32x4* gbl_i32x4_ptr;
typedef __attribute__((address_space(3))) i32x4* lds_i32x4_ptr;
#else
#define USE_ASYNC_LDS 0
#endif

__device__ __forceinline__ void slab_wait_async() {
#if USE_ASYNC_LDS
  __builtin_amdgcn_s_wait_asynccnt(0);
#endif
}

__device__ __forceinline__ void slab_copy(const char* __restrict__ gsrc,
                                          float4* __restrict__ ldst, int tid) {
#if USE_ASYNC_LDS
  for (int i = tid; i < SLAB_VEC4; i += 256) {
    __builtin_amdgcn_global_load_async_to_lds_b128(
        (gbl_i32x4_ptr)(gsrc + (size_t)i * 16),
        (lds_i32x4_ptr)(ldst + i), 0, 0);
  }
#else
  const float4* src = (const float4*)gsrc;
  for (int i = tid; i < SLAB_VEC4; i += 256) ldst[i] = src[i];
#endif
}

// ---------------- prep kernels ----------------

// Build zero-padded per-step weight slabs (f16) + per-site weight pairs (f32).
// step m uses weights[:, m(m+1) : m(m+1)+2m] for theta_c and the next 2 cols for theta_s.
__global__ void rbm_prep_slabs(const float* __restrict__ W, char* __restrict__ gslab) {
  const int m = blockIdx.x;      // 0..31
  const int h = threadIdx.x;     // 0..255
  const int start = m * (m + 1);
  _Float16* wp = (_Float16*)(gslab + (size_t)m * SLAB_BYTES);
  float2*  wsp = (float2*)(gslab + (size_t)m * SLAB_BYTES + 32768);
  const float* wrow = W + (size_t)h * 1056 + start;
  for (int j = 0; j < 64; ++j) {
    float w = (j < 2 * m) ? wrow[j] : 0.0f;
    wp[h * 64 + j] = (_Float16)w;
  }
  float2 s;
  s.x = wrow[2 * m];
  s.y = wrow[2 * m + 1];
  wsp[h] = s;
}

// Pack x (+/-1 ints) into per-row occupancy bitmasks.
__global__ void rbm_prep_xbits(const int* __restrict__ x,
                               unsigned long long* __restrict__ xb) {
  const int b = blockIdx.x * 256 + threadIdx.x;
  const int* row = x + (size_t)b * SORB;
  unsigned long long bits = 0ull;
  for (int j = 0; j < SORB; ++j)
    bits |= (unsigned long long)((row[j] + 1) >> 1) << j;
  xb[b] = bits;
}

// ---------------- main kernel ----------------
// grid (BATCH/128, SPLIT), 256 threads (8 waves). Each wave: 16 batch rows.
// blockIdx.y selects an 8-step range; partial probabilities multiplied later.
__global__ __launch_bounds__(256) void rbm_steps(
    const char* __restrict__ gslab,
    const unsigned long long* __restrict__ xb,
    const float* __restrict__ bias,
    float* __restrict__ partial) {
  __shared__ float4 smem[2 * SLAB_VEC4];   // double-buffered step slab

  const int tid  = threadIdx.x;
  const int lane = tid & 31;
  const int wv   = tid >> 5;
  const int hh   = lane >> 4;   // lane half: selects K block (A) / M block (C)
  const int lc   = lane & 15;
  const int m0   = blockIdx.y * STEPS_PER;
  const int rowbase = (blockIdx.x * 8 + wv) * 16;

  // ---- A fragments (16x32 f16, rows = rowbase+lc), contiguous k order (same as B):
  // half 0: elements 0-15 -> K 0..15 ; half 1: K 16..31
  const unsigned long long abits = xb[rowbase + lc];
  v16h a0, a1;
#pragma unroll
  for (int i = 0; i < 16; ++i) {
    const int j = 16 * hh + i;
    a0[i] = (_Float16)(float)((abits >> j) & 1ULL);
    a1[i] = (_Float16)(float)((abits >> (32 + j)) & 1ULL);
  }

  // ---- per-row state: bitmask, prefix counts at step m0, partial prob
  unsigned long long rb[8];
  int nu[8], nd[8];
  float prob[8];
  const unsigned long long EM = 0x5555555555555555ULL;
  const unsigned long long pre = (1ULL << (2 * m0)) - 1ULL;   // m0<=24 -> safe
#pragma unroll
  for (int r = 0; r < 8; ++r) {
    rb[r]   = xb[rowbase + 8 * hh + r];
    nu[r]   = __popcll(rb[r] & EM & pre);
    nd[r]   = __popcll(rb[r] & (EM << 1) & pre);
    prob[r] = 1.0f;
  }

  slab_copy(gslab + (size_t)m0 * SLAB_BYTES, smem, tid);
  for (int mm = 0; mm < STEPS_PER; ++mm) {
    const int m = m0 + mm;
    slab_wait_async();                                 // our async stores to LDS done
    __syncthreads();                                   // slab m ready; prior compute done
    if (mm + 1 < STEPS_PER)
      slab_copy(gslab + (size_t)(m + 1) * SLAB_BYTES, smem + ((mm + 1) & 1) * SLAB_VEC4, tid);

    const _Float16* wl  = (const _Float16*)(smem + (mm & 1) * SLAB_VEC4);
    const float2*   wsl = (const float2*)((const char*)(smem + (mm & 1) * SLAB_VEC4) + 32768);

    float pa[8][4];
#pragma unroll
    for (int r = 0; r < 8; ++r) {
      pa[r][0] = 1.0f; pa[r][1] = 1.0f; pa[r][2] = 1.0f; pa[r][3] = 1.0f;
    }

    // 16 HID tiles of 16; each lane owns column h = 16n+lc of the C tile
    for (int n = 0; n < 16; ++n) {
      const int h  = 16 * n + lc;
      const int k0 = 16 * hh;   // contiguous k as in A: half selects K 0..15 vs 16..31
      v16h b0 = *(const v16h*)(wl + h * 64 + k0);
      v16h b1 = *(const v16h*)(wl + h * 64 + 32 + k0);
      v8f c = {0.f, 0.f, 0.f, 0.f, 0.f, 0.f, 0.f, 0.f};
      c = __builtin_amdgcn_wmma_f32_16x16x32_f16(false, a0, false, b0, (short)0, c, false, false);
      c = __builtin_amdgcn_wmma_f32_16x16x32_f16(false, a1, false, b1, (short)0, c, false, false);

      const float2 wsv = wsl[h];
      const float cw0 = __cosf(wsv.x), sw0 = __sinf(wsv.x);
      const float cw1 = __cosf(wsv.y), sw1 = __sinf(wsv.y);
      const float cwb = cw0 * cw1 - sw0 * sw1;
      const float swb = sw0 * cw1 + cw0 * sw1;
      const float bn  = bias[h];
#pragma unroll
      for (int r = 0; r < 8; ++r) {
        const float a  = c[r] + bn;                  // C layout: VGPR r -> M = r + 8*hh
        const float ca = __cosf(a), sa = __sinf(a);  // angle addition for the 4 states
        pa[r][0] *= ca;
        pa[r][1] *= ca * cw0 - sa * sw0;
        pa[r][2] *= ca * cw1 - sa * sw1;
        pa[r][3] *= ca * cwb - sa * swb;
      }
    }

    // product over hidden units: lanes 0..15 (and 16..31) hold disjoint h sets;
    // xor-butterfly within each 16-lane half completes the 256-term product.
#pragma unroll
    for (int off = 1; off < 16; off <<= 1) {
#pragma unroll
      for (int r = 0; r < 8; ++r) {
#pragma unroll
        for (int s = 0; s < 4; ++s)
          pa[r][s] *= __shfl_xor(pa[r][s], off, 32);
      }
    }

    // per-row normalize / mask / select (mirrors reference exactly)
#pragma unroll
    for (int r = 0; r < 8; ++r) {
      const int b0i = (int)((rb[r] >> (2 * m)) & 1ULL);
      const int b1i = (int)((rb[r] >> (2 * m + 1)) & 1ULL);
      const int st  = b0i + 2 * b1i;
      float v0 = pa[r][0], v1 = pa[r][1], v2 = pa[r][2], v3 = pa[r][3];
      float n1 = fmaxf(sqrtf(v0 * v0 + v1 * v1 + v2 * v2 + v3 * v3), 1e-12f);
      v0 /= n1; v1 /= n1; v2 /= n1; v3 /= n1;
      if (m >= 8) {                                   // alpha_e <= k
        const bool ou = nu[r] < 16, uu = nu[r] > m - 16;
        const bool od = nd[r] < 16, ud = nd[r] > m - 16;
        v0 = (uu && ud) ? v0 : 0.0f;
        v1 = (ou && ud) ? v1 : 0.0f;
        v2 = (uu && od) ? v2 : 0.0f;
        v3 = (ou && od) ? v3 : 0.0f;
        float n2 = fmaxf(sqrtf(v0 * v0 + v1 * v1 + v2 * v2 + v3 * v3), 1e-12f);
        v0 /= n2; v1 /= n2; v2 /= n2; v3 /= n2;
      }
      const float sel = (st == 0) ? v0 : (st == 1) ? v1 : (st == 2) ? v2 : v3;
      prob[r] *= sel;
      nu[r] += b0i;
      nd[r] += b1i;
    }
  }

  // results are replicated across the 16 lanes of each half; lane lc==0 stores
  if (lc == 0) {
    float* dst = partial + (size_t)blockIdx.y * BATCH + rowbase + 8 * hh;
#pragma unroll
    for (int r = 0; r < 8; ++r) dst[r] = prob[r];
  }
}

// ---------------- combine ----------------
__global__ void rbm_combine(const float* __restrict__ partial, float* __restrict__ out) {
  const int b = blockIdx.x * 256 + threadIdx.x;
  float p = 1.0f;
#pragma unroll
  for (int s = 0; s < SPLIT; ++s) p *= partial[(size_t)s * BATCH + b];
  out[b] = p;
}

// ---------------- launch ----------------
extern "C" void kernel_launch(void* const* d_in, const int* in_sizes, int n_in,
                              void* d_out, int out_size, void* d_ws, size_t ws_size,
                              hipStream_t stream) {
  const int*   x       = (const int*)d_in[0];
  const float* weights = (const float*)d_in[1];
  const float* bias    = (const float*)d_in[2];

  char* ws = (char*)d_ws;
  char* gslab = ws;                                             // 32 * 34816 = 1,114,112 B
  unsigned long long* xbits =
      (unsigned long long*)(ws + 32 * (size_t)SLAB_BYTES);      // 64 KB
  float* partial =
      (float*)(ws + 32 * (size_t)SLAB_BYTES + (size_t)BATCH * 8); // SPLIT*BATCH floats

  rbm_prep_slabs<<<NSTEP, HID, 0, stream>>>(weights, gslab);
  rbm_prep_xbits<<<BATCH / 256, 256, 0, stream>>>(x, xbits);
  rbm_steps<<<dim3(BATCH / 128, SPLIT), 256, 0, stream>>>(gslab, xbits, bias, partial);
  rbm_combine<<<BATCH / 256, 256, 0, stream>>>(partial, (float*)d_out);
}
